// HierarchicalAutoregressivePolicy_42545946034677
// MI455X (gfx1250) — compile-verified, matrix operand order checked by reading, not run on hardware
//
#include <hip/hip_runtime.h>
#include <hip/hip_bf16.h>
#include <cstdint>
#include <cstddef>

// ---------------------------------------------------------------------------
// HierarchicalAutoregressivePolicy for MI455X (gfx1250, wave32, WMMA)
//
// d_in layout (jax pytree: dicts flattened in sorted-key order):
//  0 initial_images (32,4,2,3,224,224) f32
//  1 initial_states (32,4,8) f32
//  2 language_tokens (32,16) i32
//  3 W_arm 4 W_bwd 5 W_dec 6 W_fwd 7 W_goal 8 W_grip 9 W_imgp 10 W_lang
//  11 W_patch 12 W_prog 13 W_sp 14 bN 15 b_arm 16 b_bwd 17 b_dec 18 b_fwd
//  19 b_goal 20 b_grip 21 b_imgp 22 b_lang 23 b_patch 24 b_prog 25 b_sp
//  26 bwd_q 27 fwd_q 28 gN 29 goal_q 30 lang_emb
//  31 L.W1 32 L.W2 33 L.Wk 34 L.Wo 35 L.Wq 36 L.Wv
//  37 L.b1 38 L.b2 39 L.be1 40 L.be2 41 L.bk 42 L.bo 43 L.bq 44 L.bv
//  45 L.g1 46 L.g2 47 pos 48 prog_q
// ---------------------------------------------------------------------------

typedef __attribute__((ext_vector_type(16))) _Float16 v16h;
typedef __attribute__((ext_vector_type(8)))  _Float16 v8h;
typedef __attribute__((ext_vector_type(8)))  float    v8f;

#define BSZ   32
#define SEQ   530
#define DIM   1024
#define NHEAD 16
#define DH    64
#define NTOK  (BSZ * SEQ)      // 16960, multiple of 32
#define DFF_  4096
#define SPAD  544              // 34 key tiles * 16

__device__ __forceinline__ v8f wmma16(v16h a, v16h b, v8f c) {
  return __builtin_amdgcn_wmma_f32_16x16x32_f16(
      /*neg_a=*/false, a, /*neg_b=*/false, b,
      /*c_mod=*/(short)0, c, /*reuse_a=*/false, /*reuse_b=*/false);
}

// K-offset pattern for 16-bit A/B fragments (16x32 ISA VGPR layout):
//   lane<16: K = i + (i&8);   lane>=16: K = 8 + i + (i&8)
// => per lane the fragment is two contiguous 8-half (16B) chunks at
//    base+kb and base+kb+16. Load as two b128s.
__device__ __forceinline__ int kofs(int lane, int i) {
  return ((lane & 16) ? 8 : 0) + i + (i & 8);
}

// p must already include the per-lane kb offset (0 or 8 halves) and be 16B aligned
__device__ __forceinline__ v16h load_frag(const _Float16* __restrict__ p) {
  const v8h lo = *(const v8h*)(p);
  const v8h hi = *(const v8h*)(p + 16);
  return __builtin_shufflevector(lo, hi, 0, 1, 2, 3, 4, 5, 6, 7,
                                 8, 9, 10, 11, 12, 13, 14, 15);
}

// ---------------------------------------------------------------------------
// WMMA GEMM:  C[M,N] = act(A[M,K] @ W[K,N] + bias), with W given TRANSPOSED
// as Wt[N,K] so B-fragments are contiguous-in-K (pure b128 loads).
// A,Wt row-major f16, 16B aligned. K%32==0, M%32==0, N arbitrary (clamped:
// out-of-range columns are computed on a clamped address, never stored).
// One wave computes a 32(M) x 64(N) strip: 2 A-frags + 4 B-frags per K-step
// (12 b128 loads) feed 8 WMMAs; all fragments are loaded into distinct
// registers before the WMMA burst so loadcnt waits overlap.
// act: 0=none 1=gelu(exact erf) 2=sigmoid.
// ---------------------------------------------------------------------------
__global__ __launch_bounds__(256)
void gemm_f16_kernel(const _Float16* __restrict__ A,
                     const _Float16* __restrict__ Wt,
                     const float* __restrict__ bias,
                     float* __restrict__ Cf,
                     _Float16* __restrict__ Ch,
                     int M, int N, int K, int act) {
  const int lane = threadIdx.x & 31;
  const int wave = blockIdx.x * (blockDim.x >> 5) + (threadIdx.x >> 5);
  const int ntiles = (N + 63) >> 6;
  const int mtiles = M >> 5;                    // 32-row strips
  if (wave >= mtiles * ntiles) return;          // uniform per-wave exit
  const int mt = wave / ntiles;
  const int nt = wave % ntiles;
  const int m  = lane & 15;
  const int kb = (lane & 16) ? 8 : 0;
  const int rhalf = (lane & 16) ? 8 : 0;

  const _Float16* __restrict__ Ap0 = A + (size_t)(mt * 32 + m) * K + kb;
  const _Float16* __restrict__ Ap1 = Ap0 + (size_t)16 * K;
  const _Float16* __restrict__ Bp[4];
#pragma unroll
  for (int j = 0; j < 4; ++j) {
    int col = nt * 64 + j * 16 + m;
    if (col >= N) col = N - 1;                  // clamp: extra cols never stored
    Bp[j] = Wt + (size_t)col * K + kb;
  }

  v8f acc[2][4] = {{v8f{}, v8f{}, v8f{}, v8f{}},
                   {v8f{}, v8f{}, v8f{}, v8f{}}};
  for (int k0 = 0; k0 < K; k0 += 32) {
    // issue all 12 b128 loads first (distinct regs), then the WMMA burst
    const v16h a0 = load_frag(Ap0 + k0);
    const v16h a1 = load_frag(Ap1 + k0);
    v16h bf[4];
#pragma unroll
    for (int j = 0; j < 4; ++j) bf[j] = load_frag(Bp[j] + k0);
    if (k0 + 32 < K) {
      __builtin_prefetch(Ap0 + k0 + 32, 0, 1);
      __builtin_prefetch(Ap1 + k0 + 32, 0, 1);
    }
#pragma unroll
    for (int j = 0; j < 4; ++j) {
      acc[0][j] = wmma16(a0, bf[j], acc[0][j]);
      acc[1][j] = wmma16(a1, bf[j], acc[1][j]);
    }
  }

#pragma unroll
  for (int j = 0; j < 4; ++j) {
    const int col = nt * 64 + j * 16 + m;
    if (col >= N) continue;
    const float bs = bias ? bias[col] : 0.0f;
#pragma unroll
    for (int si = 0; si < 2; ++si) {
#pragma unroll
      for (int r = 0; r < 8; ++r) {
        const int row = mt * 32 + si * 16 + r + rhalf;
        float v = acc[si][j][r] + bs;
        if (act == 1)      v = 0.5f * v * (1.0f + erff(v * 0.70710678118f));
        else if (act == 2) v = 1.0f / (1.0f + __expf(-v));
        if (Cf) Cf[(size_t)row * N + col] = v;
        if (Ch) Ch[(size_t)row * N + col] = (_Float16)v;
      }
    }
  }
}

// ---------------------------------------------------------------------------
// Causal attention. One wave per (q-tile, head, batch).
// Scores via WMMA into LDS (f32), softmax in LDS, ctx = P @ V via WMMA.
// ---------------------------------------------------------------------------
__global__ __launch_bounds__(32)
void attn_kernel(const _Float16* __restrict__ Q,
                 const _Float16* __restrict__ Km,
                 const _Float16* __restrict__ Vm,
                 _Float16* __restrict__ Ctx) {
  __shared__ float sc[16][SPAD];
  const int lane = threadIdx.x;
  const int qt = blockIdx.x, h = blockIdx.y, b = blockIdx.z;
  const int m = lane & 15;
  const int kb = (lane & 16) ? 8 : 0;
  const int rhalf = (lane & 16) ? 8 : 0;

  // zero score pad (keys beyond written range must read as prob 0)
  for (int i = lane; i < 16 * SPAD; i += 32) (&sc[0][0])[i] = 0.0f;
  __syncthreads();

  // Q fragments: dh=64 -> two K=32 chunks, contiguous per lane
  int qrow = qt * 16 + m; if (qrow >= SEQ) qrow = SEQ - 1;
  const _Float16* Qp = Q + ((size_t)(b * SEQ + qrow)) * DIM + h * DH + kb;
  const v16h qa0 = load_frag(Qp);
  const v16h qa1 = load_frag(Qp + 32);

  for (int nt = 0; nt <= qt; ++nt) {
    int kcol = nt * 16 + m; if (kcol >= SEQ) kcol = SEQ - 1;
    const _Float16* Kp = Km + ((size_t)(b * SEQ + kcol)) * DIM + h * DH + kb;
    const v16h b0 = load_frag(Kp);
    const v16h b1 = load_frag(Kp + 32);
    v8f acc = v8f{};
    acc = wmma16(qa0, b0, acc);
    acc = wmma16(qa1, b1, acc);
#pragma unroll
    for (int r = 0; r < 8; ++r) sc[r + rhalf][nt * 16 + m] = acc[r];
  }
  __syncthreads();

  // softmax: lanes 0..15 each own a row
  if (lane < 16) {
    const int qg = qt * 16 + lane;
    const int ncol = (qt + 1) * 16;
    const float scale = 0.125f;              // 1/sqrt(64)
    float mx = -3.0e38f;
    for (int c = 0; c < ncol; ++c) {
      const bool ok = (c <= qg) && (c < SEQ) && (qg < SEQ);
      if (ok) mx = fmaxf(mx, sc[lane][c] * scale);
    }
    float sum = 0.0f;
    for (int c = 0; c < ncol; ++c) {
      const bool ok = (c <= qg) && (c < SEQ) && (qg < SEQ);
      const float e = ok ? __expf(sc[lane][c] * scale - mx) : 0.0f;
      sc[lane][c] = e; sum += e;
    }
    const float inv = (sum > 0.0f) ? 1.0f / sum : 0.0f;
    for (int c = 0; c < ncol; ++c) sc[lane][c] *= inv;
  }
  __syncthreads();

  // ctx = P(16 x keys) @ V(keys x 64); 4 N-subtiles of 16
  const int nchunks = ((qt + 1) * 16 + 31) >> 5;
  v8f acc[4] = {v8f{}, v8f{}, v8f{}, v8f{}};
  for (int c = 0; c < nchunks; ++c) {
    const int k0 = c * 32;
    v16h pa;
#pragma unroll
    for (int i = 0; i < 16; ++i) pa[i] = (_Float16)sc[m][k0 + kofs(lane, i)];
#pragma unroll
    for (int j = 0; j < 4; ++j) {
      v16h bv;
#pragma unroll
      for (int i = 0; i < 16; ++i) {
        int key = k0 + kofs(lane, i); if (key >= SEQ) key = SEQ - 1;
        bv[i] = Vm[((size_t)(b * SEQ + key)) * DIM + h * DH + j * 16 + m];
      }
      acc[j] = wmma16(pa, bv, acc[j]);
    }
  }
#pragma unroll
  for (int j = 0; j < 4; ++j)
#pragma unroll
    for (int r = 0; r < 8; ++r) {
      const int row = qt * 16 + r + rhalf;
      if (row < SEQ)
        Ctx[((size_t)(b * SEQ + row)) * DIM + h * DH + j * 16 + m] =
            (_Float16)acc[j][r];
    }
}

// ---------------------------------------------------------------------------
// Fused residual + LayerNorm (D=1024, 256 threads/row). dx may be null.
// ---------------------------------------------------------------------------
__global__ __launch_bounds__(256)
void resid_ln_kernel(float* __restrict__ x, const float* __restrict__ dx,
                     const float* __restrict__ g, const float* __restrict__ be,
                     _Float16* __restrict__ xh) {
  __shared__ float s1[256], s2[256];
  const int tid = threadIdx.x;
  const size_t row = blockIdx.x;
  float* xr = x + row * DIM;
  const float* dr = dx ? dx + row * DIM : nullptr;

  float v[4], a = 0.0f, b = 0.0f;
#pragma unroll
  for (int i = 0; i < 4; ++i) {
    const int c = tid + i * 256;
    float t = xr[c] + (dr ? dr[c] : 0.0f);
    v[i] = t; a += t; b += t * t;
  }
  s1[tid] = a; s2[tid] = b;
  __syncthreads();
  for (int o = 128; o > 0; o >>= 1) {
    if (tid < o) { s1[tid] += s1[tid + o]; s2[tid] += s2[tid + o]; }
    __syncthreads();
  }
  const float mean = s1[0] * (1.0f / DIM);
  const float var  = s2[0] * (1.0f / DIM) - mean * mean;
  const float inv  = rsqrtf(var + 1e-5f);
#pragma unroll
  for (int i = 0; i < 4; ++i) {
    const int c = tid + i * 256;
    const float y = (v[i] - mean) * inv * g[c] + be[c];
    xr[c] = y;
    xh[row * DIM + c] = (_Float16)y;
  }
}

// ---------------------------------------------------------------------------
// Small data-movement / embed kernels
// ---------------------------------------------------------------------------

// Transposing f32->f16 weight convert: in = L stacked (K,N) matrices,
// out = L stacked (N,K) matrices. Writes coalesced along K.
__global__ void f2h_t_kernel(const float* __restrict__ in,
                             _Float16* __restrict__ out,
                             int K, int N, long total) {
  const long idx = (long)blockIdx.x * 256 + threadIdx.x;
  if (idx >= total) return;
  const long kn = (long)K * N;
  const int  l  = (int)(idx / kn);
  const long r  = idx - (long)l * kn;
  const int  n  = (int)(r / K);
  const int  k  = (int)(r - (long)n * K);
  out[idx] = (_Float16)in[(size_t)l * kn + (size_t)k * N + n];
}

// rows: r = b*512 + o*128 + cam*64 + latent ; cols: 3*16*16 patch pixels
__global__ void patch_gather_kernel(const float* __restrict__ img,
                                    _Float16* __restrict__ P) {
  const size_t idx = (size_t)blockIdx.x * 256 + threadIdx.x;
  if (idx >= (size_t)16384 * 768) return;
  const int k = (int)(idx % 768);
  const int r = (int)(idx / 768);
  const int b = r >> 9, rem = r & 511;
  const int o = rem >> 7, c = (rem >> 6) & 1, l = rem & 63;
  const int gr = l / 14, gc = l % 14;
  const int ch = k >> 8, pr = (k >> 4) & 15, pc = k & 15;
  const size_t off = ((((size_t)b * 4 + o) * 2 + c) * 3 + ch) * (224 * 224) +
                     (size_t)(gr * 16 + pr) * 224 + (gc * 16 + pc);
  P[idx] = (_Float16)img[off];
}

__global__ void lang_mean_kernel(const int* __restrict__ tok,
                                 const float* __restrict__ emb,
                                 _Float16* __restrict__ out) {
  const int idx = blockIdx.x * 256 + threadIdx.x;
  if (idx >= 32 * 512) return;
  const int b = idx >> 9, d = idx & 511;
  float s = 0.0f;
  for (int t = 0; t < 16; ++t) s += emb[(size_t)tok[b * 16 + t] * 512 + d];
  out[idx] = (_Float16)(s * (1.0f / 16.0f));
}

// cat[r,0:1024]=states[:6]@W_arm+b_arm ; cat[r,1024:]=states[6:]@W_grip+b_grip
__global__ void state_cat_kernel(const float* __restrict__ st,
                                 const float* __restrict__ Wa, const float* __restrict__ ba,
                                 const float* __restrict__ Wg, const float* __restrict__ bg,
                                 _Float16* __restrict__ out) {
  const int idx = blockIdx.x * 256 + threadIdx.x;
  if (idx >= 128 * 2048) return;
  const int r = idx >> 11, j = idx & 2047;
  const float* s = st + r * 8;
  float v;
  if (j < 1024) {
    v = ba[j];
    for (int i = 0; i < 6; ++i) v += s[i] * Wa[i * 1024 + j];
  } else {
    const int jj = j - 1024;
    v = bg[jj];
    for (int i = 0; i < 2; ++i) v += s[6 + i] * Wg[i * 1024 + jj];
  }
  out[idx] = (_Float16)v;
}

__global__ void assemble_kernel(const float* __restrict__ lang,
                                const float* __restrict__ state,
                                const float* __restrict__ enc,
                                const float* __restrict__ progq,
                                const float* __restrict__ goalq,
                                const float* __restrict__ bwdq,
                                const float* __restrict__ fwdq,
                                const float* __restrict__ pos,
                                float* __restrict__ x, _Float16* __restrict__ xh) {
  const size_t idx = (size_t)blockIdx.x * 256 + threadIdx.x;
  if (idx >= (size_t)NTOK * DIM) return;
  const int d = (int)(idx & (DIM - 1));
  const size_t t = idx >> 10;
  const int b = (int)(t / SEQ), s = (int)(t % SEQ);
  float v;
  if (s == 0)        v = lang[b * DIM + d];
  else if (s < 5)    v = state[(b * 4 + (s - 1)) * DIM + d];
  else if (s < 517)  v = enc[((size_t)b * 512 + (s - 5)) * DIM + d];
  else if (s == 517) v = progq[d];
  else if (s < 522)  v = goalq[(s - 518) * DIM + d];
  else if (s < 526)  v = bwdq[(s - 522) * DIM + d];
  else               v = fwdq[(s - 526) * DIM + d];
  v += pos[s * DIM + d];
  x[idx] = v;
  xh[idx] = (_Float16)v;
}

// pooled queries -> h128: [0:32)=prog, [32:64)=goal, [64:96)=bwd, [96:128)=fwd
__global__ void heads_kernel(const float* __restrict__ x, _Float16* __restrict__ h) {
  const int idx = blockIdx.x * 256 + threadIdx.x;
  if (idx >= 128 * DIM) return;
  const int d = idx & (DIM - 1), r = idx >> 10;
  const int b = r & 31, kind = r >> 5;
  const float* base = x + (size_t)b * SEQ * DIM;
  float v;
  if (kind == 0) v = base[(size_t)517 * DIM + d];
  else {
    const int s0 = 518 + (kind - 1) * 4;
    v = 0.25f * (base[(size_t)s0 * DIM + d] + base[(size_t)(s0 + 1) * DIM + d] +
                 base[(size_t)(s0 + 2) * DIM + d] + base[(size_t)(s0 + 3) * DIM + d]);
  }
  h[idx] = (_Float16)v;
}

// ---------------------------------------------------------------------------
// Host launcher
// ---------------------------------------------------------------------------
static inline void launch_gemm(hipStream_t st, const _Float16* A, const _Float16* Wt,
                               const float* bias, float* Cf, _Float16* Ch,
                               int M, int N, int K, int act) {
  const int strips = (M / 32) * ((N + 63) / 64);
  gemm_f16_kernel<<<(strips + 7) / 8, 256, 0, st>>>(A, Wt, bias, Cf, Ch, M, N, K, act);
}

extern "C" void kernel_launch(void* const* d_in, const int* in_sizes, int n_in,
                              void* d_out, int out_size, void* d_ws, size_t ws_size,
                              hipStream_t stream) {
  (void)in_sizes; (void)n_in; (void)out_size; (void)ws_size;

  const float* images = (const float*)d_in[0];
  const float* states = (const float*)d_in[1];
  const int*   toks   = (const int*)d_in[2];
  const float* W_arm  = (const float*)d_in[3];
  const float* W_bwd  = (const float*)d_in[4];
  const float* W_dec  = (const float*)d_in[5];
  const float* W_fwd  = (const float*)d_in[6];
  const float* W_goal = (const float*)d_in[7];
  const float* W_grip = (const float*)d_in[8];
  const float* W_imgp = (const float*)d_in[9];
  const float* W_lang = (const float*)d_in[10];
  const float* W_patch= (const float*)d_in[11];
  const float* W_prog = (const float*)d_in[12];
  const float* W_sp   = (const float*)d_in[13];
  const float* bN     = (const float*)d_in[14];
  const float* b_arm  = (const float*)d_in[15];
  const float* b_bwd  = (const float*)d_in[16];
  const float* b_dec  = (const float*)d_in[17];
  const float* b_fwd  = (const float*)d_in[18];
  const float* b_goal = (const float*)d_in[19];
  const float* b_grip = (const float*)d_in[20];
  const float* b_imgp = (const float*)d_in[21];
  const float* b_lang = (const float*)d_in[22];
  const float* b_patch= (const float*)d_in[23];
  const float* b_prog = (const float*)d_in[24];
  const float* b_sp   = (const float*)d_in[25];
  const float* bwd_q  = (const float*)d_in[26];
  const float* fwd_q  = (const float*)d_in[27];
  const float* gN     = (const float*)d_in[28];
  const float* goal_q = (const float*)d_in[29];
  const float* lang_emb = (const float*)d_in[30];
  const float* L_W1  = (const float*)d_in[31];
  const float* L_W2  = (const float*)d_in[32];
  const float* L_Wk  = (const float*)d_in[33];
  const float* L_Wo  = (const float*)d_in[34];
  const float* L_Wq  = (const float*)d_in[35];
  const float* L_Wv  = (const float*)d_in[36];
  const float* L_b1  = (const float*)d_in[37];
  const float* L_b2  = (const float*)d_in[38];
  const float* L_be1 = (const float*)d_in[39];
  const float* L_be2 = (const float*)d_in[40];
  const float* L_bk  = (const float*)d_in[41];
  const float* L_bo  = (const float*)d_in[42];
  const float* L_bq  = (const float*)d_in[43];
  const float* L_bv  = (const float*)d_in[44];
  const float* L_g1  = (const float*)d_in[45];
  const float* L_g2  = (const float*)d_in[46];
  const float* pos   = (const float*)d_in[47];
  const float* prog_q= (const float*)d_in[48];

  float* out = (float*)d_out;

  // ---- workspace bump allocator ----
  char* cur = (char*)d_ws;
  auto alloc = [&](size_t bytes) -> void* {
    void* p = (void*)cur;
    cur += (bytes + 255) & ~(size_t)255;
    return p;
  };
  // transposing converts: in = L x (K,N) f32 -> out = L x (N,K) f16
  auto f2h_t = [&](const float* src, _Float16* dst, int K, int N, int L) {
    const long total = (long)L * K * N;
    f2h_t_kernel<<<(unsigned)((total + 255) / 256), 256, 0, stream>>>(src, dst, K, N, total);
  };

  // f16 transposed weights
  _Float16* Wpatch_h = (_Float16*)alloc((size_t)768 * 768 * 2);
  _Float16* Wimgp_h  = (_Float16*)alloc((size_t)768 * 1024 * 2);
  _Float16* Wlang_h  = (_Float16*)alloc((size_t)512 * 1024 * 2);
  _Float16* Wsp_h    = (_Float16*)alloc((size_t)2048 * 1024 * 2);
  _Float16* Wq_h = (_Float16*)alloc((size_t)8 * 1024 * 1024 * 2);
  _Float16* Wk_h = (_Float16*)alloc((size_t)8 * 1024 * 1024 * 2);
  _Float16* Wv_h = (_Float16*)alloc((size_t)8 * 1024 * 1024 * 2);
  _Float16* Wo_h = (_Float16*)alloc((size_t)8 * 1024 * 1024 * 2);
  _Float16* W1_h = (_Float16*)alloc((size_t)8 * 1024 * 4096 * 2);
  _Float16* W2_h = (_Float16*)alloc((size_t)8 * 4096 * 1024 * 2);
  _Float16* Wgoal_h = (_Float16*)alloc((size_t)1024 * 768 * 2);
  _Float16* Wdec_h  = (_Float16*)alloc((size_t)768 * 3072 * 2);
  _Float16* Wbwd_h  = (_Float16*)alloc((size_t)1024 * 32 * 2);
  _Float16* Wfwd_h  = (_Float16*)alloc((size_t)1024 * 32 * 2);
  _Float16* Wprog_h = (_Float16*)alloc((size_t)1024 * 1 * 2);
  // activations
  _Float16* P_h    = (_Float16*)alloc((size_t)16384 * 768 * 2);
  _Float16* enc1_h = (_Float16*)alloc((size_t)16384 * 768 * 2);
  float*    enc2_f = (float*)   alloc((size_t)16384 * 1024 * 4);
  _Float16* lm_h   = (_Float16*)alloc((size_t)32 * 512 * 2);
  float*    lang_f = (float*)   alloc((size_t)32 * 1024 * 4);
  _Float16* cat_h  = (_Float16*)alloc((size_t)128 * 2048 * 2);
  float*    state_f= (float*)   alloc((size_t)128 * 1024 * 4);
  float*    x_f    = (float*)   alloc((size_t)NTOK * DIM * 4);
  _Float16* x_h    = (_Float16*)alloc((size_t)NTOK * DIM * 2);
  _Float16* q_h    = (_Float16*)alloc((size_t)NTOK * DIM * 2);
  _Float16* k_h    = (_Float16*)alloc((size_t)NTOK * DIM * 2);
  _Float16* v_h    = (_Float16*)alloc((size_t)NTOK * DIM * 2);
  _Float16* ctx_h  = (_Float16*)alloc((size_t)NTOK * DIM * 2);
  float*    proj_f = (float*)   alloc((size_t)NTOK * DIM * 4);
  _Float16* ff_h   = (_Float16*)alloc((size_t)NTOK * DFF_ * 2);
  _Float16* h128_h = (_Float16*)alloc((size_t)128 * 1024 * 2);
  _Float16* lat_h  = (_Float16*)alloc((size_t)32 * 768 * 2);

  // ---- weight conversion + transpose ----
  f2h_t(W_patch, Wpatch_h, 768, 768, 1);
  f2h_t(W_imgp,  Wimgp_h,  768, 1024, 1);
  f2h_t(W_lang,  Wlang_h,  512, 1024, 1);
  f2h_t(W_sp,    Wsp_h,    2048, 1024, 1);
  f2h_t(L_Wq, Wq_h, 1024, 1024, 8);
  f2h_t(L_Wk, Wk_h, 1024, 1024, 8);
  f2h_t(L_Wv, Wv_h, 1024, 1024, 8);
  f2h_t(L_Wo, Wo_h, 1024, 1024, 8);
  f2h_t(L_W1, W1_h, 1024, 4096, 8);
  f2h_t(L_W2, W2_h, 4096, 1024, 8);
  f2h_t(W_goal, Wgoal_h, 1024, 768, 1);
  f2h_t(W_dec,  Wdec_h,  768, 3072, 1);
  f2h_t(W_bwd,  Wbwd_h,  1024, 32, 1);
  f2h_t(W_fwd,  Wfwd_h,  1024, 32, 1);
  f2h_t(W_prog, Wprog_h, 1024, 1, 1);

  // ---- patch embed: (16384,768) @ Wpatch -> @ Wimgp ----
  patch_gather_kernel<<<(16384u * 768u) / 256u, 256, 0, stream>>>(images, P_h);
  launch_gemm(stream, P_h,    Wpatch_h, b_patch, nullptr, enc1_h, 16384, 768, 768, 0);
  launch_gemm(stream, enc1_h, Wimgp_h,  b_imgp,  enc2_f,  nullptr, 16384, 1024, 768, 0);

  // ---- language / state embeds ----
  lang_mean_kernel<<<(32 * 512 + 255) / 256, 256, 0, stream>>>(toks, lang_emb, lm_h);
  launch_gemm(stream, lm_h, Wlang_h, b_lang, lang_f, nullptr, 32, 1024, 512, 0);
  state_cat_kernel<<<(128 * 2048 + 255) / 256, 256, 0, stream>>>(
      states, W_arm, b_arm, W_grip, b_grip, cat_h);
  launch_gemm(stream, cat_h, Wsp_h, b_sp, state_f, nullptr, 128, 1024, 2048, 0);

  // ---- assemble x = concat(...) + pos ----
  assemble_kernel<<<(unsigned)(((size_t)NTOK * DIM + 255) / 256), 256, 0, stream>>>(
      lang_f, state_f, enc2_f, prog_q, goal_q, bwd_q, fwd_q, pos, x_f, x_h);

  // ---- transformer layers ----
  const dim3 agrid((SEQ + 15) / 16, NHEAD, BSZ);
  for (int l = 0; l < 8; ++l) {
    const size_t wo = (size_t)l * 1024 * 1024;
    const size_t wf = (size_t)l * 1024 * 4096;
    launch_gemm(stream, x_h, Wq_h + wo, L_bq + l * 1024, nullptr, q_h, NTOK, DIM, DIM, 0);
    launch_gemm(stream, x_h, Wk_h + wo, L_bk + l * 1024, nullptr, k_h, NTOK, DIM, DIM, 0);
    launch_gemm(stream, x_h, Wv_h + wo, L_bv + l * 1024, nullptr, v_h, NTOK, DIM, DIM, 0);
    attn_kernel<<<agrid, 32, 0, stream>>>(q_h, k_h, v_h, ctx_h);
    launch_gemm(stream, ctx_h, Wo_h + wo, L_bo + l * 1024, proj_f, nullptr, NTOK, DIM, DIM, 0);
    resid_ln_kernel<<<NTOK, 256, 0, stream>>>(x_f, proj_f, L_g1 + l * 1024,
                                              L_be1 + l * 1024, x_h);
    launch_gemm(stream, x_h, W1_h + wf, L_b1 + l * 4096, nullptr, ff_h, NTOK, DFF_, DIM, 1);
    launch_gemm(stream, ff_h, W2_h + wf, L_b2 + l * 1024, proj_f, nullptr, NTOK, DIM, DFF_, 0);
    resid_ln_kernel<<<NTOK, 256, 0, stream>>>(x_f, proj_f, L_g2 + l * 1024,
                                              L_be2 + l * 1024, x_h);
  }

  // ---- final LN (in place, no residual) ----
  resid_ln_kernel<<<NTOK, 256, 0, stream>>>(x_f, nullptr, gN, bN, x_h);

  // ---- heads ----
  heads_kernel<<<(128 * 1024 + 255) / 256, 256, 0, stream>>>(x_f, h128_h);
  launch_gemm(stream, h128_h + (size_t)32 * 1024, Wgoal_h, b_goal, nullptr, lat_h, 32, 768, 1024, 0);
  launch_gemm(stream, lat_h, Wdec_h, b_dec, out + 2048, nullptr, 32, 3072, 768, 0);
  launch_gemm(stream, h128_h + (size_t)64 * 1024, Wbwd_h, b_bwd, out,        nullptr, 32, 32, 1024, 0);
  launch_gemm(stream, h128_h + (size_t)96 * 1024, Wfwd_h, b_fwd, out + 1024, nullptr, 32, 32, 1024, 0);
  launch_gemm(stream, h128_h, Wprog_h, b_prog, out + 100352, nullptr, 32, 1, 1024, 2);
}
